// MultiScaleAttention_1159641170063
// MI455X (gfx1250) — matrix operand order repaired and checked by reading, not compiled, not run on hardware
//
#include <hip/hip_runtime.h>

// ---------------------------------------------------------------------------
// MultiScaleAttention for MI455X (gfx1250): all GEMMs via v_wmma_f32_16x16x32_bf16
// wave32; A-frag: lane holds row=lane%16, K = (lane/16)*16 + j
//          B-frag: lane holds col=lane%16, K = (lane/16)*16 + j
//          C/D   : lane holds col=lane%16, rows (lane/16)*8 + r
// Software-pipelined (double-buffered) fragment loads so WMMAs overlap VMEM.
// ---------------------------------------------------------------------------

typedef __attribute__((ext_vector_type(16))) __bf16 v16bf;
typedef __attribute__((ext_vector_type(8)))  float  v8f;

#define BATCH   4
#define SEQ     2048
#define DMODEL  1024
#define NSC     4
#define EMB     256
#define HEADS   4
#define DKH     64

__device__ __forceinline__ unsigned short f2bf(float f) {
  union { float f; unsigned u; } v; v.f = f;
  unsigned r = (v.u + 0x7FFFu + ((v.u >> 16) & 1u)) >> 16;
  return (unsigned short)r;
}

#define WMMA_BF16(A, B, C) \
  __builtin_amdgcn_wmma_f32_16x16x32_bf16(false, (A), false, (B), (short)0, (C), false, false)

// ---------------------------- elementwise helpers --------------------------
__global__ void cvt_f32_bf16(const float* __restrict__ src,
                             unsigned short* __restrict__ dst, long n) {
  long i = (long)blockIdx.x * blockDim.x + threadIdx.x;
  if (i < n) dst[i] = f2bf(src[i]);
}

__global__ void transpose_cvt(const float* __restrict__ src,   // (R,C) row-major
                              unsigned short* __restrict__ dst, // (C,R) row-major
                              int R, int C) {
  long i = (long)blockIdx.x * blockDim.x + threadIdx.x;
  if (i >= (long)R * C) return;
  int r = (int)(i / C), c = (int)(i % C);
  dst[(long)c * R + r] = f2bf(src[i]);
}

__global__ void zero_f32(float* __restrict__ p, long n) {
  long i = (long)blockIdx.x * blockDim.x + threadIdx.x;
  if (i < n) p[i] = 0.0f;
}

// ---------------------------- generic bf16 WMMA GEMM -----------------------
// C(M,N) = A(M,K) @ BT(N,K)^T + bias.  A rows remapped:
//   off(r) = (r / rows_per_batch)*batch_stride + (r % rows_per_batch)*row_stride
// One wave computes a 16x64 tile; K-loop is double-buffered (K % 64 == 0).
// blockDim=256 (8 waves), grid=(N/64, M/128).
__global__ void gemm_bf16(const unsigned short* __restrict__ A,
                          const unsigned short* __restrict__ BT,
                          const float* __restrict__ bias,
                          unsigned short* __restrict__ Cbf,  // bf16 out (or null)
                          float* __restrict__ Cf,            // f32 out (or null)
                          int M, int N, int K, int ldb, int ldc,
                          int rows_per_batch, long row_stride, long batch_stride) {
  int lane = threadIdx.x & 31, wv = threadIdx.x >> 5;
  int row0 = (blockIdx.y * 8 + wv) * 16;
  int col0 = blockIdx.x * 64;
  if (row0 >= M || col0 >= N) return;
  int half = lane >> 4, lr = lane & 15;

  const __bf16* Ab = (const __bf16*)A;
  const __bf16* Bb = (const __bf16*)BT;

  int arow = row0 + lr;
  long aoff = (long)(arow / rows_per_batch) * batch_stride +
              (long)(arow % rows_per_batch) * row_stride;
  long bcol[4];
#pragma unroll
  for (int n = 0; n < 4; ++n) bcol[n] = (long)(col0 + n * 16 + lr) * ldb;

  auto loadAB = [&](v16bf& af, v16bf (&bf)[4], int k0) {
    const __bf16* ap = Ab + aoff + k0 + half * 16;
    __builtin_prefetch(ap + 256, 0, 1);   // global_prefetch_b8 eight k-steps ahead
#pragma unroll
    for (int j = 0; j < 8; ++j) { af[j] = ap[j]; af[8 + j] = ap[8 + j]; }
#pragma unroll
    for (int n = 0; n < 4; ++n) {
      const __bf16* bp = Bb + bcol[n] + k0 + half * 16;
#pragma unroll
      for (int j = 0; j < 16; ++j) bf[n][j] = bp[j];
    }
  };

  v8f acc[4] = {};
  v16bf aA, aB; v16bf bA[4], bB[4];
  int nk = K >> 5;                        // even for all our shapes
  loadAB(aA, bA, 0);
  for (int ki = 0; ki < nk; ki += 2) {
    loadAB(aB, bB, (ki + 1) << 5);
#pragma unroll
    for (int n = 0; n < 4; ++n) acc[n] = WMMA_BF16(aA, bA[n], acc[n]);
    if (ki + 2 < nk) loadAB(aA, bA, (ki + 2) << 5);
#pragma unroll
    for (int n = 0; n < 4; ++n) acc[n] = WMMA_BF16(aB, bB[n], acc[n]);
  }

#pragma unroll
  for (int n = 0; n < 4; ++n) {
    int col = col0 + n * 16 + lr;
    float bv = bias ? bias[col] : 0.0f;
#pragma unroll
    for (int r = 0; r < 8; ++r) {
      int row = row0 + half * 8 + r;
      float v = acc[n][r] + bv;
      long o = (long)row * ldc + col;
      if (Cf) Cf[o] = v; else Cbf[o] = f2bf(v);
    }
  }
}

// -------------------- v^T extraction: qkv(B*L,768) -> vT(B,H,64,L) ---------
__global__ void extract_vT(const unsigned short* __restrict__ qkv,
                           unsigned short* __restrict__ vT, int L) {
  long i = (long)blockIdx.x * blockDim.x + threadIdx.x;
  long total = (long)BATCH * HEADS * DKH * L;
  if (i >= total) return;
  int l = (int)(i % L); long t = i / L;
  int dk = (int)(t % DKH); t /= DKH;
  int h = (int)(t % HEADS); int b = (int)(t / HEADS);
  vT[i] = qkv[((long)b * L + l) * 768 + 512 + h * DKH + dk];
}

// -------------------- pass 1: online softmax stats (m, l) per row ----------
// grid = (L/64, B*H), block = 128 (4 waves; 1 wave = 16 q rows)
// Key-tile fragments double-buffered: shuffle-reduction overlaps next loads.
__global__ void attn_stats(const unsigned short* __restrict__ qkv,
                           float* __restrict__ stats, int L) {
  int lane = threadIdx.x & 31, wv = threadIdx.x >> 5;
  int bh = blockIdx.y, b = bh >> 2, h = bh & 3;
  int row0 = (blockIdx.x * 4 + wv) * 16;
  int half = lane >> 4, lr = lane & 15;
  const __bf16* Q = (const __bf16*)qkv;

  v16bf a0, a1;
  {
    const __bf16* ap = Q + ((long)(b * L + row0 + lr)) * 768 + h * DKH + half * 16;
#pragma unroll
    for (int j = 0; j < 8; ++j) {
      a0[j] = ap[j]; a0[8 + j] = ap[8 + j];
      a1[j] = ap[32 + j]; a1[8 + j] = ap[40 + j];
    }
  }
  float m[8], l[8];
#pragma unroll
  for (int r = 0; r < 8; ++r) { m[r] = -1e30f; l[r] = 0.0f; }

  auto loadK = [&](v16bf& b0, v16bf& b1, int kt) {
    const __bf16* kp = Q + ((long)(b * L + kt * 16 + lr)) * 768 + 256 + h * DKH + half * 16;
#pragma unroll
    for (int j = 0; j < 16; ++j) { b0[j] = kp[j]; b1[j] = kp[32 + j]; }
  };
  auto reduceTile = [&](const v8f& d) {
#pragma unroll
    for (int r = 0; r < 8; ++r) {
      float s = d[r] * 0.125f;            // 1/sqrt(DK)
      float tm = s;
#pragma unroll
      for (int off = 1; off < 16; off <<= 1) tm = fmaxf(tm, __shfl_xor(tm, off, 32));
      float nm = fmaxf(m[r], tm);
      float p = __expf(s - nm);
#pragma unroll
      for (int off = 1; off < 16; off <<= 1) p += __shfl_xor(p, off, 32);
      l[r] = l[r] * __expf(m[r] - nm) + p;
      m[r] = nm;
    }
  };

  v16bf b0A, b1A, b0B, b1B;
  int nkt = L >> 4;                       // always even (L >= 256)
  loadK(b0A, b1A, 0);
  for (int kt = 0; kt < nkt; kt += 2) {
    loadK(b0B, b1B, kt + 1);
    v8f d = {};
    d = WMMA_BF16(a0, b0A, d);
    d = WMMA_BF16(a1, b1A, d);
    reduceTile(d);
    if (kt + 2 < nkt) loadK(b0A, b1A, kt + 2);
    v8f e = {};
    e = WMMA_BF16(a0, b0B, e);
    e = WMMA_BF16(a1, b1B, e);
    reduceTile(e);
  }
  if (lr == 0) {
#pragma unroll
    for (int r = 0; r < 8; ++r) {
      int row = row0 + half * 8 + r;
      stats[((long)bh * L + row) * 2]     = m[r];
      stats[((long)bh * L + row) * 2 + 1] = l[r];
    }
  }
}

// -------------------- pass 2: o = softmax(qk^T) v  (+ w0 for scale 0) ------
__global__ void attn_av(const unsigned short* __restrict__ qkv,
                        const unsigned short* __restrict__ vT,
                        const float* __restrict__ stats,
                        unsigned short* __restrict__ attnO,
                        float* __restrict__ w0, int L) {
  __shared__ unsigned short ptile[4][16 * 32];
  int lane = threadIdx.x & 31, wv = threadIdx.x >> 5;
  int bh = blockIdx.y, b = bh >> 2, h = bh & 3;
  int row0 = (blockIdx.x * 4 + wv) * 16;
  int half = lane >> 4, lr = lane & 15;
  const __bf16* Q = (const __bf16*)qkv;

  v16bf a0, a1;
  {
    const __bf16* ap = Q + ((long)(b * L + row0 + lr)) * 768 + h * DKH + half * 16;
#pragma unroll
    for (int j = 0; j < 8; ++j) {
      a0[j] = ap[j]; a0[8 + j] = ap[8 + j];
      a1[j] = ap[32 + j]; a1[8 + j] = ap[40 + j];
    }
  }
  float m[8], li[8];
#pragma unroll
  for (int r = 0; r < 8; ++r) {
    int row = row0 + half * 8 + r;
    m[r] = stats[((long)bh * L + row) * 2];
    float lv = stats[((long)bh * L + row) * 2 + 1];
    li[r] = (lv > 0.0f) ? 1.0f / lv : 0.0f;
  }
  v8f acc[4] = {};
  unsigned short* lds = ptile[wv];
  const __bf16* ldsr = (const __bf16*)ptile[wv];
  const __bf16* Vb = (const __bf16*)vT;

  for (int kt = 0; kt < (L >> 5); ++kt) {
    // Load both key subtiles' fragments up front.
    v16bf kb0[2], kb1[2];
#pragma unroll
    for (int t = 0; t < 2; ++t) {
      const __bf16* kp =
          Q + ((long)(b * L + kt * 32 + t * 16 + lr)) * 768 + 256 + h * DKH + half * 16;
#pragma unroll
      for (int j = 0; j < 16; ++j) { kb0[t][j] = kp[j]; kb1[t][j] = kp[32 + j]; }
    }
    // Load v fragments BEFORE the LDS barrier so they overlap the softmax work.
    v16bf bv[4];
#pragma unroll
    for (int n = 0; n < 4; ++n) {
      const __bf16* vp = Vb + ((long)bh * 64 + n * 16 + lr) * L + kt * 32 + half * 16;
#pragma unroll
      for (int j = 0; j < 16; ++j) bv[n][j] = vp[j];
    }
    // Scores -> normalized probs -> LDS (D-layout tile) [+ w0 for scale 0].
#pragma unroll
    for (int t = 0; t < 2; ++t) {
      v8f d = {};
      d = WMMA_BF16(a0, kb0[t], d);
      d = WMMA_BF16(a1, kb1[t], d);
      int key0 = kt * 32 + t * 16;
#pragma unroll
      for (int r = 0; r < 8; ++r) {
        float p = __expf(d[r] * 0.125f - m[r]) * li[r];
        int rit = half * 8 + r;
        lds[rit * 32 + t * 16 + lr] = f2bf(p);
        if (w0)  // scale 0: w0 += p / HEADS
          atomicAdd(&w0[((long)(b * L) + row0 + rit) * L + key0 + lr], 0.25f * p);
      }
    }
    // LDS relayout: D-fragment (row-major tile) -> A-fragment for next WMMA.
    asm volatile("s_wait_dscnt 0" ::: "memory");
    v16bf pa;
    {
      int kb = half * 16;
#pragma unroll
      for (int j = 0; j < 8; ++j) {
        pa[j]     = ldsr[lr * 32 + kb + j];
        pa[8 + j] = ldsr[lr * 32 + 8 + kb + j];
      }
    }
#pragma unroll
    for (int n = 0; n < 4; ++n) acc[n] = WMMA_BF16(pa, bv[n], acc[n]);
  }
#pragma unroll
  for (int n = 0; n < 4; ++n) {
#pragma unroll
    for (int r = 0; r < 8; ++r) {
      int row = row0 + half * 8 + r;
      int col = h * DKH + n * 16 + lr;
      attnO[((long)(b * L) + row) * 256 + col] = f2bf(acc[n][r]);
    }
  }
}

// -------------------- upsample (linear interp) + concat into z (bf16) ------
__global__ void upsample_concat(const float* __restrict__ y, // (B*L, 256) f32
                                unsigned short* __restrict__ zbf,
                                int L, int s, int colOff) {
  long i = (long)blockIdx.x * blockDim.x + threadIdx.x;
  long total = (long)BATCH * SEQ * EMB;
  if (i >= total) return;
  int d = (int)(i % EMB); long t = i / EMB;
  int lpos = (int)(t % SEQ); int b = (int)(t / SEQ);
  int k = lpos / s;
  float frac = (float)(lpos % s) / (float)s;
  long base = ((long)b * L + k) * EMB + d;
  float y0 = y[base];
  float y1 = (k + 1 < L) ? y[base + EMB] : 0.0f;
  zbf[((long)b * SEQ + lpos) * DMODEL + colOff + d] = f2bf((1.0f - frac) * y0 + frac * y1);
}

// -------------------- LayerNorm over 1024, one block per row ---------------
__global__ void layernorm(const float* __restrict__ z, const float* __restrict__ g,
                          const float* __restrict__ bb, float* __restrict__ out) {
  int row = blockIdx.x;
  const float* zr = z + (long)row * DMODEL;
  float s = 0.0f, s2 = 0.0f;
  for (int c = threadIdx.x; c < DMODEL; c += 256) { float v = zr[c]; s += v; s2 += v * v; }
#pragma unroll
  for (int off = 16; off; off >>= 1) { s += __shfl_xor(s, off, 32); s2 += __shfl_xor(s2, off, 32); }
  __shared__ float red[2][8]; __shared__ float smu, srs;
  int lane = threadIdx.x & 31, wv = threadIdx.x >> 5;
  if (lane == 0) { red[0][wv] = s; red[1][wv] = s2; }
  __syncthreads();
  if (threadIdx.x == 0) {
    float ts = 0.0f, ts2 = 0.0f;
    for (int w = 0; w < 8; ++w) { ts += red[0][w]; ts2 += red[1][w]; }
    float mu = ts / (float)DMODEL;
    float var = ts2 / (float)DMODEL - mu * mu;
    smu = mu; srs = rsqrtf(var + 1e-5f);
  }
  __syncthreads();
  float mu = smu, rs = srs;
  for (int c = threadIdx.x; c < DMODEL; c += 256)
    out[(long)row * DMODEL + c] = (zr[c] - mu) * rs * g[c] + bb[c];
}

// ---------------------------------------------------------------------------
extern "C" void kernel_launch(void* const* d_in, const int* in_sizes, int n_in,
                              void* d_out, int out_size, void* d_ws, size_t ws_size,
                              hipStream_t stream) {
  const float* x      = (const float*)d_in[0];
  const float* proj_W = (const float*)d_in[1];
  const float* proj_b = (const float*)d_in[2];
  const float* in_W   = (const float*)d_in[3];
  const float* in_b   = (const float*)d_in[4];
  const float* out_W  = (const float*)d_in[5];
  const float* out_b  = (const float*)d_in[6];
  const float* fus_W  = (const float*)d_in[7];
  const float* fus_b  = (const float*)d_in[8];
  const float* ln_g   = (const float*)d_in[9];
  const float* ln_bb  = (const float*)d_in[10];

  char* ws = (char*)d_ws;
  size_t off = 0;
  auto alloc = [&](size_t bytes) -> char* {
    off = (off + 255) & ~(size_t)255;
    char* p = ws + off; off += bytes; return p;
  };
  unsigned short* xbf    = (unsigned short*)alloc(8192UL * 1024 * 2);
  unsigned short* projWT = (unsigned short*)alloc(4UL * 256 * 1024 * 2);
  unsigned short* inWT   = (unsigned short*)alloc(4UL * 768 * 256 * 2);
  unsigned short* outWT  = (unsigned short*)alloc(4UL * 256 * 256 * 2);
  unsigned short* fusWT  = (unsigned short*)alloc(1024UL * 1024 * 2);
  unsigned short* xs     = (unsigned short*)alloc(8192UL * 256 * 2);
  unsigned short* qkvb   = (unsigned short*)alloc(8192UL * 768 * 2);
  unsigned short* vTb    = (unsigned short*)alloc(16UL * 64 * 2048 * 2);
  float*          stats  = (float*)alloc(16UL * 2048 * 2 * 4);
  unsigned short* attnO  = (unsigned short*)alloc(8192UL * 256 * 2);
  float*          ybuf   = (float*)alloc(8192UL * 256 * 4);
  unsigned short* zbf    = (unsigned short*)alloc(8192UL * 1024 * 2);
  float*          zf     = (float*)alloc(8192UL * 1024 * 4);
  (void)ws_size; (void)in_sizes; (void)n_in; (void)out_size;

  float* fused = (float*)d_out;
  float* w0 = fused + (long)BATCH * SEQ * DMODEL;          // (4,2048,2048)

  long nx = 8192L * 1024;
  cvt_f32_bf16<<<(int)((nx + 255) / 256), 256, 0, stream>>>(x, xbf, nx);
  for (int i = 0; i < NSC; ++i) {
    transpose_cvt<<<(1024 * 256) / 256, 256, 0, stream>>>(proj_W + (long)i * 1024 * 256,
                                                          projWT + (long)i * 256 * 1024, 1024, 256);
    transpose_cvt<<<(256 * 768) / 256, 256, 0, stream>>>(in_W + (long)i * 256 * 768,
                                                         inWT + (long)i * 768 * 256, 256, 768);
    transpose_cvt<<<(256 * 256) / 256, 256, 0, stream>>>(out_W + (long)i * 256 * 256,
                                                         outWT + (long)i * 256 * 256, 256, 256);
  }
  transpose_cvt<<<(1024 * 1024) / 256, 256, 0, stream>>>(fus_W, fusWT, 1024, 1024);

  long nw0 = (long)BATCH * SEQ * SEQ;
  zero_f32<<<(int)((nw0 + 255) / 256), 256, 0, stream>>>(w0, nw0);

  const int scales[NSC] = {1, 2, 4, 8};
  for (int i = 0; i < NSC; ++i) {
    int s = scales[i], L = SEQ / s, M = BATCH * L;
    // projection on subsampled rows: (M,1024)@(1024,256)
    gemm_bf16<<<dim3(256 / 64, M / 128), 256, 0, stream>>>(
        xbf, projWT + (long)i * 256 * 1024, proj_b + i * 256, xs, nullptr,
        M, 256, 1024, 1024, 256, L, (long)s * 1024, (long)SEQ * 1024);
    // qkv: (M,256)@(256,768)
    gemm_bf16<<<dim3(768 / 64, M / 128), 256, 0, stream>>>(
        xs, inWT + (long)i * 768 * 256, in_b + i * 768, qkvb, nullptr,
        M, 768, 256, 256, 768, M, 256L, 0L);
    long nv = (long)BATCH * HEADS * DKH * L;
    extract_vT<<<(int)((nv + 255) / 256), 256, 0, stream>>>(qkvb, vTb, L);
    attn_stats<<<dim3(L / 64, BATCH * HEADS), 128, 0, stream>>>(qkvb, stats, L);
    attn_av<<<dim3(L / 64, BATCH * HEADS), 128, 0, stream>>>(
        qkvb, vTb, stats, attnO, (i == 0) ? w0 : nullptr, L);
    // out projection: (M,256)@(256,256) -> f32
    gemm_bf16<<<dim3(256 / 64, M / 128), 256, 0, stream>>>(
        attnO, outWT + (long)i * 256 * 256, out_b + i * 256, nullptr, ybuf,
        M, 256, 256, 256, 256, M, 256L, 0L);
    long nu = (long)BATCH * SEQ * EMB;
    upsample_concat<<<(int)((nu + 255) / 256), 256, 0, stream>>>(ybuf, zbf, L, s, i * EMB);
  }
  // fusion: (8192,1024)@(1024,1024) -> f32
  gemm_bf16<<<dim3(1024 / 64, 8192 / 128), 256, 0, stream>>>(
      zbf, fusWT, fus_b, nullptr, zf, 8192, 1024, 1024, 1024, 1024, 8192, 1024L, 0L);
  layernorm<<<8192, 256, 0, stream>>>(zf, ln_g, ln_bb, fused);
}
